// ResNetV1_52115133170151
// MI455X (gfx1250) — matrix operand order repaired and checked by reading, not compiled, not run on hardware
//
#include <hip/hip_runtime.h>
#include <hip/hip_bf16.h>

// ---- vector types for WMMA fragments ----------------------------------------
typedef _Float16 h16 __attribute__((ext_vector_type(16)));  // A/B fragment (8 VGPRs)
typedef _Float16 h8  __attribute__((ext_vector_type(8)));   // 16B load chunk
typedef float    v8f __attribute__((ext_vector_type(8)));   // C/D fragment

#define NCH   64                    // channels
#define KOFF  27                    // 3x3x3 offsets
#define WFRAG (KOFF * 2 * 4 * 32 * 16) // 110592 f16 per packed weight tensor

// -----------------------------------------------------------------------------
// Pack W[k][c][d] (fp32, 27x64x64) into f16 B-fragment layout:
//   frag(k, kk, nt): 32 lanes x 16 halves, lane -> N = nt*16 + (lane&15),
//   halves h -> K(row c) = kk*32 + (lane>=16)*16 + h.
// Each lane's 16 halves are contiguous (32B) -> B load = straight b128 pair.
// -----------------------------------------------------------------------------
__global__ __launch_bounds__(256) void sconv_pack_w(
    const float* __restrict__ W0, const float* __restrict__ W1,
    _Float16* __restrict__ wh0, _Float16* __restrict__ wh1) {
  int r = blockIdx.x * blockDim.x + threadIdx.x;
  if (r >= 2 * WFRAG) return;
  const float* W   = (r < WFRAG) ? W0  : W1;
  _Float16*    dst = (r < WFRAG) ? wh0 : wh1;
  unsigned q = (r < WFRAG) ? r : r - WFRAG;
  unsigned hh   =  q        & 15;
  unsigned lane = (q >> 4)  & 31;
  unsigned nt   = (q >> 9)  & 3;
  unsigned kk   = (q >> 11) & 1;
  unsigned k    =  q >> 12;
  unsigned c = kk * 32 + ((lane >> 4) << 4) + hh;   // K dim (input channel)
  unsigned d = nt * 16 + (lane & 15);               // N dim (output channel)
  dst[q] = (_Float16)W[(k * NCH + c) * NCH + d];
}

// fp32 -> f16 feature conversion (4 elements/thread)
__global__ __launch_bounds__(256) void sconv_cvt_x(
    const float* __restrict__ x, _Float16* __restrict__ xh, long total4) {
  long i = (long)blockIdx.x * blockDim.x + threadIdx.x;
  if (i >= total4) return;
  const float4 v = ((const float4*)x)[i];
  _Float16* o = xh + i * 4;
  o[0] = (_Float16)v.x; o[1] = (_Float16)v.y;
  o[2] = (_Float16)v.z; o[3] = (_Float16)v.w;
}

// -----------------------------------------------------------------------------
// One sparse-conv layer. Wave = 64 rows (4 M-tiles of 16) x all 64 channels.
//   RES=false: out_h = relu(conv)            (f16, to workspace)
//   RES=true : out_f = relu(conv + resid)    (f32, to d_out)
// All per-lane addressing uses 32-bit unsigned byte/element offsets against a
// uniform (SGPR) base so loads lower to the saddr (GVS) form: one VALU per
// address instead of 64-bit shift+add chains.
// -----------------------------------------------------------------------------
template <bool RES>
__global__ __launch_bounds__(128) void sconv_layer(
    const _Float16* __restrict__ feat,   // [n,64] f16
    const int*      __restrict__ nbr,    // [n,27] int32
    const _Float16* __restrict__ wh,     // packed f16 weights (WFRAG)
    const float*    __restrict__ resid,  // [n,64] f32 (RES only)
    float*          __restrict__ outf,   // RES
    _Float16*       __restrict__ outh,   // !RES
    int n) {
  const int lane = threadIdx.x & 31;
  const int wave = threadIdx.x >> 5;
  const int waveBase = (blockIdx.x * 4 + wave) * 64;
  if (waveBase >= n) return;

  const unsigned l15 = lane & 15;
  const unsigned hi  = lane >> 4;        // 0 | 1
  const unsigned sub = hi << 5;          // byte offset of channel sub-block (s=16 halves)

  v8f acc[4][4] = {};                    // [M-tile][N-tile], 128 VGPRs f32 accum

  const char* featB = (const char*)feat;
  const char* nbrB  = (const char*)nbr;
  const _Float16* whl = wh + lane * 16;  // per-lane weight fragment base

  // per-tile neighbor-table element base (hoisted out of k loop), 32-bit
  unsigned ib[4];
  #pragma unroll
  for (int t = 0; t < 4; ++t) {
    int nidx = waveBase + t * 16 + (int)l15;
    nidx = nidx < n ? nidx : n - 1;      // clamp: EXEC stays all-1 for WMMA
    ib[t] = (unsigned)nidx * (KOFF * 4u);   // byte offset of nbr[nidx][0]
  }

  #pragma unroll 1
  for (int k = 0; k < KOFF; ++k) {
    // ---- gather A fragments: 4 tiles x 2 K-chunks, contiguous k = 16*hi + s order ----
    h16 a[4][2];
    #pragma unroll
    for (int t = 0; t < 4; ++t) {
      const int row = *(const int*)(nbrB + (ib[t] + (unsigned)k * 4u));
      const unsigned off = ((unsigned)row << 7) | sub;  // row*128B + 32B*hi
      h8 p0 = *(const h8*)(featB + off);        // ch  s..s+7
      h8 p1 = *(const h8*)(featB + off + 16);   // ch  8+s..15+s
      h8 p2 = *(const h8*)(featB + off + 64);   // ch  32+s..39+s
      h8 p3 = *(const h8*)(featB + off + 80);   // ch  40+s..47+s
      a[t][0] = __builtin_shufflevector(p0, p1, 0,1,2,3,4,5,6,7,8,9,10,11,12,13,14,15);
      a[t][1] = __builtin_shufflevector(p2, p3, 0,1,2,3,4,5,6,7,8,9,10,11,12,13,14,15);
    }
    // ---- 32 WMMAs: B reused across the 4 M-tiles --------------------------
    const _Float16* wk = whl + k * 4096;     // k * 2*4*512
    #pragma unroll
    for (int kk = 0; kk < 2; ++kk) {
      #pragma unroll
      for (int nt = 0; nt < 4; ++nt) {
        h16 b = *(const h16*)(wk + (kk * 4 + nt) * 512);
        #pragma unroll
        for (int t = 0; t < 4; ++t) {
          acc[t][nt] = __builtin_amdgcn_wmma_f32_16x16x32_f16(
              false, a[t][kk], false, b, (short)0, acc[t][nt], false, false);
        }
      }
    }
  }

  // ---- epilogue: C/D layout -> rows; relu (+residual) -----------------------
  const char* residB = (const char*)resid;
  char* outfB = (char*)outf;
  char* outhB = (char*)outh;
  #pragma unroll
  for (int t = 0; t < 4; ++t) {
    const int rbase = waveBase + t * 16;
    if (rbase < n) {                       // n % 16 == 0 -> tile-uniform guard
      #pragma unroll
      for (int j = 0; j < 8; ++j) {
        const unsigned m = (unsigned)rbase + hi * 8 + (unsigned)j;   // row (M)
        const unsigned e = m * NCH + l15;  // element index of (m, l15)
        #pragma unroll
        for (int nt = 0; nt < 4; ++nt) {
          float v = acc[t][nt][j];
          const unsigned el = e + nt * 16u;           // (m, nt*16+l15)
          if (RES) {
            v += *(const float*)(residB + el * 4u);
            *(float*)(outfB + el * 4u) = fmaxf(v, 0.f);
          } else {
            *(_Float16*)(outhB + el * 2u) = (_Float16)fmaxf(v, 0.f);
          }
        }
      }
    }
  }
}

// -----------------------------------------------------------------------------
extern "C" void kernel_launch(void* const* d_in, const int* in_sizes, int n_in,
                              void* d_out, int out_size, void* d_ws, size_t ws_size,
                              hipStream_t stream) {
  const float* x   = (const float*)d_in[0];
  const int*   nbr = (const int*)d_in[1];   // JAX x64 off -> int32 indices
  const float* W0  = (const float*)d_in[2];
  const float* W1  = (const float*)d_in[3];
  float* out = (float*)d_out;
  const int n = in_sizes[0] / NCH;          // 150000

  // workspace: [wh0 | wh1 | xh (f16) | h (f16)]  ~= 0.43MB + 2*19.2MB
  _Float16* wh0 = (_Float16*)d_ws;
  _Float16* wh1 = wh0 + WFRAG;
  _Float16* xh  = wh1 + WFRAG;
  _Float16* hbf = xh + (size_t)n * NCH;

  sconv_pack_w<<<(2 * WFRAG + 255) / 256, 256, 0, stream>>>(W0, W1, wh0, wh1);
  const long total4 = (long)n * NCH / 4;
  sconv_cvt_x<<<(int)((total4 + 255) / 256), 256, 0, stream>>>(x, xh, total4);

  const int blocks = (n + 255) / 256;       // 4 waves/block, 64 rows/wave
  sconv_layer<false><<<blocks, 128, 0, stream>>>(xh,  nbr, wh0, nullptr, nullptr, hbf, n);
  sconv_layer<true ><<<blocks, 128, 0, stream>>>(hbf, nbr, wh1, x,       out,     nullptr, n);
}
